// BaseVectorQuantizer_9861244911832
// MI455X (gfx1250) — matrix operand order repaired and checked by reading, not compiled, not run on hardware
//
#include <hip/hip_runtime.h>
#include <hip/hip_bf16.h>

typedef __attribute__((ext_vector_type(16))) __bf16 v16bf;
typedef __attribute__((ext_vector_type(8)))  float  v8f;
typedef unsigned short u16;

#define LN_EPS 1e-5f

union Frag16 { uint4 q[2]; v16bf v; };

static __device__ __forceinline__ v16bf make_frag(uint4 a, uint4 b) {
  Frag16 f; f.q[0] = a; f.q[1] = b; return f.v;
}

// ---------------------------------------------------------------------------
// f32 -> bf16 hi/lo split (bf16x3 decomposition)
// ---------------------------------------------------------------------------
__global__ void split_bf16_kernel(const float* __restrict__ src,
                                  __bf16* __restrict__ hi,
                                  __bf16* __restrict__ lo, int n) {
  int i = blockIdx.x * blockDim.x + threadIdx.x;
  if (i < n) {
    float f = src[i];
    __bf16 h = (__bf16)f;
    hi[i] = h;
    lo[i] = (__bf16)(f - (float)h);
  }
}

// ---------------------------------------------------------------------------
// ||emb_k||^2 per codebook row (block of 256 threads per row, D == 256)
// ---------------------------------------------------------------------------
__global__ void rownorm_kernel(const float* __restrict__ emb,
                               float* __restrict__ embn, int D) {
  int k = blockIdx.x;
  float e = emb[(size_t)k * D + threadIdx.x];
  float s = e * e;
  #pragma unroll
  for (int m = 16; m >= 1; m >>= 1) s += __shfl_xor(s, m, 32);
  __shared__ float part[8];
  int wave = threadIdx.x >> 5;
  if ((threadIdx.x & 31) == 0) part[wave] = s;
  __syncthreads();
  if (threadIdx.x == 0) {
    float t = 0.f;
    #pragma unroll
    for (int w = 0; w < 8; w++) t += part[w];
    embn[k] = t;
  }
}

// ---------------------------------------------------------------------------
// Fused projection (features @ Wp^T + bp) + LayerNorm, output bf16 hi/lo.
// Block: 512 threads = 16 waves. Row block = 16, each wave owns one 16-col
// tile of D=256. bf16x3 via v_wmma_f32_16x16x32_bf16.
// ---------------------------------------------------------------------------
__global__ __launch_bounds__(512)
void proj_ln_kernel(const __bf16* __restrict__ fh, const __bf16* __restrict__ fl,
                    const __bf16* __restrict__ wph, const __bf16* __restrict__ wpl,
                    const float* __restrict__ bp, const float* __restrict__ gamma,
                    const float* __restrict__ beta,
                    __bf16* __restrict__ xh, __bf16* __restrict__ xl,
                    int H, int D) {
  const int n0   = blockIdx.x * 16;
  const int tid  = threadIdx.x;
  const int wave = tid >> 5;      // 0..15 -> d-tile
  const int lane = tid & 31;
  const int hf   = lane >> 4;     // half of wave
  const int lcol = lane & 15;     // column within 16x16 tile

  __shared__ __align__(16) u16 sAh[16][520];   // padded stride: bank-conflict free
  __shared__ __align__(16) u16 sAl[16][520];
  __shared__ float sPartS[16][16];
  __shared__ float sPartQ[16][16];
  __shared__ float sMu[16], sRs[16];

  // Cooperative load of 16 x H feature rows (hi & lo) into LDS.
  const int qperrow = H >> 3;            // uint4 per row
  for (int q = tid; q < 16 * qperrow; q += blockDim.x) {
    int r = q / qperrow, c = q % qperrow;
    *(uint4*)&sAh[r][c * 8] = ((const uint4*)(fh + (size_t)(n0 + r) * H))[c];
    *(uint4*)&sAl[r][c * 8] = ((const uint4*)(fl + (size_t)(n0 + r) * H))[c];
  }
  __syncthreads();

  const int dcol = wave * 16 + lcol;     // this lane's output column
  v8f acc = {};
  for (int h0 = 0; h0 < H; h0 += 32) {
    // A fragment: row = lcol; lane half hf covers K = hf*16 + 0..15
    v16bf ah = make_frag(*(const uint4*)&sAh[lcol][h0 + hf * 16],
                         *(const uint4*)&sAh[lcol][h0 + hf * 16 + 8]);
    v16bf al = make_frag(*(const uint4*)&sAl[lcol][h0 + hf * 16],
                         *(const uint4*)&sAl[lcol][h0 + hf * 16 + 8]);
    // B fragment from Wp[dcol][h]: lane half hf covers K = hf*16 + 0..15
    const __bf16* pbh = wph + (size_t)dcol * H + h0 + hf * 16;
    const __bf16* pbl = wpl + (size_t)dcol * H + h0 + hf * 16;
    v16bf bh = make_frag(((const uint4*)pbh)[0], ((const uint4*)pbh)[1]);
    v16bf bl = make_frag(((const uint4*)pbl)[0], ((const uint4*)pbl)[1]);
    acc = __builtin_amdgcn_wmma_f32_16x16x32_bf16(false, ah, false, bh, (short)0, acc, false, false);
    acc = __builtin_amdgcn_wmma_f32_16x16x32_bf16(false, ah, false, bl, (short)0, acc, false, false);
    acc = __builtin_amdgcn_wmma_f32_16x16x32_bf16(false, al, false, bh, (short)0, acc, false, false);
  }

  const float bias = bp[dcol];
  const float g    = gamma[dcol];
  const float bt   = beta[dcol];
  float vs[8];
  #pragma unroll
  for (int r = 0; r < 8; r++) vs[r] = acc[r] + bias;

  // Deterministic LayerNorm reduction: wave-local shuffles + LDS table.
  #pragma unroll
  for (int r = 0; r < 8; r++) {
    float s = vs[r], q = vs[r] * vs[r];
    #pragma unroll
    for (int m = 1; m < 16; m <<= 1) {
      s += __shfl_xor(s, m, 16);
      q += __shfl_xor(q, m, 16);
    }
    if (lcol == 0) {
      sPartS[r + 8 * hf][wave] = s;
      sPartQ[r + 8 * hf][wave] = q;
    }
  }
  __syncthreads();
  if (tid < 16) {
    float s = 0.f, q = 0.f;
    #pragma unroll
    for (int w = 0; w < 16; w++) { s += sPartS[tid][w]; q += sPartQ[tid][w]; }
    float mu  = s / (float)D;
    float var = q / (float)D - mu * mu;
    sMu[tid] = mu;
    sRs[tid] = rsqrtf(var + LN_EPS);
  }
  __syncthreads();

  #pragma unroll
  for (int r = 0; r < 8; r++) {
    int row = r + 8 * hf;
    float y = (vs[r] - sMu[row]) * sRs[row] * g + bt;
    __bf16 h = (__bf16)y;
    size_t off = (size_t)(n0 + row) * D + dcol;
    xh[off] = h;
    xl[off] = (__bf16)(y - (float)h);
  }
}

// ---------------------------------------------------------------------------
// Distance + argmin + gather. score_k = ||v_k||^2 - 2 u.v_k (||u||^2 dropped).
// Block: 512 threads = 16 waves; row block = 32 = two 16-row tiles.
// Register blocking: each wave owns 16 consecutive k-tiles and computes BOTH
// row tiles per B fragment (6 WMMAs per 4 B-loads), halving the emb/L2 stream.
// ---------------------------------------------------------------------------
__global__ __launch_bounds__(512)
void dist_argmin_kernel(const __bf16* __restrict__ xh, const __bf16* __restrict__ xl,
                        const __bf16* __restrict__ eh, const __bf16* __restrict__ el,
                        const float* __restrict__ embn, const float* __restrict__ emb,
                        float* __restrict__ qout, float* __restrict__ idxout,
                        int* __restrict__ idxint, int D, int K) {
  const int n0   = blockIdx.x * 32;
  const int tid  = threadIdx.x;
  const int wave = tid >> 5;      // 0..15 -> k-group
  const int lane = tid & 31;
  const int hf   = lane >> 4;
  const int lcol = lane & 15;

  __shared__ __align__(16) u16 sXh[32][264];
  __shared__ __align__(16) u16 sXl[32][264];
  __shared__ float sMin[32][16];
  __shared__ int   sIdx[32][16];
  __shared__ int   sRowIdx[32];

  const int qperrow = D >> 3;            // 32 uint4 per row
  for (int q = tid; q < 32 * qperrow; q += blockDim.x) {
    int r = q / qperrow, c = q % qperrow;
    *(uint4*)&sXh[r][c * 8] = ((const uint4*)(xh + (size_t)(n0 + r) * D))[c];
    *(uint4*)&sXl[r][c * 8] = ((const uint4*)(xl + (size_t)(n0 + r) * D))[c];
  }
  __syncthreads();

  float minv0[8], minv1[8];
  int   mini0[8], mini1[8];
  #pragma unroll
  for (int r = 0; r < 8; r++) {
    minv0[r] = __builtin_inff(); mini0[r] = 0;
    minv1[r] = __builtin_inff(); mini1[r] = 0;
  }

  const int tilesPerWave = (K / 16) / 16;   // 16 for K=4096
  for (int t = 0; t < tilesPerWave; t++) {
    const int kcol = (wave * tilesPerWave + t) * 16 + lcol;
    const __bf16* pbh = eh + (size_t)kcol * D + hf * 16;
    const __bf16* pbl = el + (size_t)kcol * D + hf * 16;
    if (t + 1 < tilesPerWave) {           // prefetch next k-tile's emb rows
      __builtin_prefetch(pbh + (size_t)16 * D, 0, 0);
      __builtin_prefetch(pbl + (size_t)16 * D, 0, 0);
    }
    v8f acc0 = {};
    v8f acc1 = {};
    #pragma unroll
    for (int d0 = 0; d0 < 256; d0 += 32) {
      // B fragments (shared by both row tiles)
      v16bf bh = make_frag(((const uint4*)(pbh + d0))[0], ((const uint4*)(pbh + d0))[1]);
      v16bf bl = make_frag(((const uint4*)(pbl + d0))[0], ((const uint4*)(pbl + d0))[1]);
      // A fragments, row tile 0 (rows 0..15) and row tile 1 (rows 16..31)
      v16bf a0h = make_frag(*(const uint4*)&sXh[lcol][d0 + hf * 16],
                            *(const uint4*)&sXh[lcol][d0 + hf * 16 + 8]);
      v16bf a0l = make_frag(*(const uint4*)&sXl[lcol][d0 + hf * 16],
                            *(const uint4*)&sXl[lcol][d0 + hf * 16 + 8]);
      v16bf a1h = make_frag(*(const uint4*)&sXh[16 + lcol][d0 + hf * 16],
                            *(const uint4*)&sXh[16 + lcol][d0 + hf * 16 + 8]);
      v16bf a1l = make_frag(*(const uint4*)&sXl[16 + lcol][d0 + hf * 16],
                            *(const uint4*)&sXl[16 + lcol][d0 + hf * 16 + 8]);
      acc0 = __builtin_amdgcn_wmma_f32_16x16x32_bf16(false, a0h, false, bh, (short)0, acc0, false, false);
      acc0 = __builtin_amdgcn_wmma_f32_16x16x32_bf16(false, a0h, false, bl, (short)0, acc0, false, false);
      acc0 = __builtin_amdgcn_wmma_f32_16x16x32_bf16(false, a0l, false, bh, (short)0, acc0, false, false);
      acc1 = __builtin_amdgcn_wmma_f32_16x16x32_bf16(false, a1h, false, bh, (short)0, acc1, false, false);
      acc1 = __builtin_amdgcn_wmma_f32_16x16x32_bf16(false, a1h, false, bl, (short)0, acc1, false, false);
      acc1 = __builtin_amdgcn_wmma_f32_16x16x32_bf16(false, a1l, false, bh, (short)0, acc1, false, false);
    }
    const float en = embn[kcol];
    #pragma unroll
    for (int r = 0; r < 8; r++) {
      float s0 = en - 2.0f * acc0[r];
      float s1 = en - 2.0f * acc1[r];
      if (s0 < minv0[r]) { minv0[r] = s0; mini0[r] = kcol; }  // ascending k: first-min
      if (s1 < minv1[r]) { minv1[r] = s1; mini1[r] = kcol; }
    }
  }

  // Cross-lane argmin within each 16-lane group (each group shares rows).
  #pragma unroll
  for (int r = 0; r < 8; r++) {
    float v0 = minv0[r]; int i0 = mini0[r];
    float v1 = minv1[r]; int i1 = mini1[r];
    #pragma unroll
    for (int m = 1; m < 16; m <<= 1) {
      float ov0 = __shfl_xor(v0, m, 16);
      int   oi0 = __shfl_xor(i0, m, 16);
      if (ov0 < v0 || (ov0 == v0 && oi0 < i0)) { v0 = ov0; i0 = oi0; }
      float ov1 = __shfl_xor(v1, m, 16);
      int   oi1 = __shfl_xor(i1, m, 16);
      if (ov1 < v1 || (ov1 == v1 && oi1 < i1)) { v1 = ov1; i1 = oi1; }
    }
    if (lcol == 0) {
      int row = r + 8 * hf;
      sMin[row][wave]      = v0;
      sIdx[row][wave]      = i0;
      sMin[16 + row][wave] = v1;
      sIdx[16 + row][wave] = i1;
    }
  }
  __syncthreads();

  if (tid < 32) {
    float v = sMin[tid][0];
    int  ix = sIdx[tid][0];
    #pragma unroll
    for (int w = 1; w < 16; w++) {     // ascending wave = ascending k ranges
      float ov = sMin[tid][w];
      int   oi = sIdx[tid][w];
      if (ov < v || (ov == v && oi < ix)) { v = ov; ix = oi; }
    }
    sRowIdx[tid] = ix;
    idxout[n0 + tid] = (float)ix;
    idxint[n0 + tid] = ix;
  }
  __syncthreads();

  // quantized = emb[idx] gather, coalesced.
  for (int p = tid; p < 32 * D; p += blockDim.x) {
    int r = p / D, col = p - r * D;
    qout[(size_t)(n0 + r) * D + col] = emb[(size_t)sRowIdx[r] * D + col];
  }
}

// ---------------------------------------------------------------------------
// One-hot encodings fill (memory-bandwidth floor of the whole op).
// ---------------------------------------------------------------------------
__global__ void encodings_kernel(const int* __restrict__ idx,
                                 float* __restrict__ enc,
                                 int kshift, long long total4) {
  long long i = (long long)blockIdx.x * blockDim.x + threadIdx.x;
  if (i >= total4) return;
  long long e0 = i << 2;
  int n   = (int)(e0 >> kshift);
  int col = (int)(e0 & ((1 << kshift) - 1));
  int target = idx[n];
  float4 v = {0.f, 0.f, 0.f, 0.f};
  if (target >= col && target < col + 4) (&v.x)[target - col] = 1.0f;
  ((float4*)enc)[i] = v;
}

// ---------------------------------------------------------------------------
extern "C" void kernel_launch(void* const* d_in, const int* in_sizes, int n_in,
                              void* d_out, int out_size, void* d_ws, size_t ws_size,
                              hipStream_t stream) {
  const float* features = (const float*)d_in[0];
  const float* Wp       = (const float*)d_in[1];
  const float* bp       = (const float*)d_in[2];
  const float* gamma    = (const float*)d_in[3];
  const float* beta     = (const float*)d_in[4];
  const float* emb      = (const float*)d_in[5];

  const int D = in_sizes[2];            // 256
  const int H = in_sizes[1] / D;        // 512
  const int N = in_sizes[0] / H;        // 16384
  const int K = in_sizes[5] / D;        // 4096

  // Workspace layout
  char* ws = (char*)d_ws;
  size_t off = 0;
  auto alloc = [&](size_t bytes) -> char* {
    char* p = ws + off;
    off += (bytes + 255) & ~(size_t)255;
    return p;
  };
  __bf16* fh  = (__bf16*)alloc((size_t)N * H * 2);
  __bf16* fl  = (__bf16*)alloc((size_t)N * H * 2);
  __bf16* wph = (__bf16*)alloc((size_t)D * H * 2);
  __bf16* wpl = (__bf16*)alloc((size_t)D * H * 2);
  __bf16* eh  = (__bf16*)alloc((size_t)K * D * 2);
  __bf16* el  = (__bf16*)alloc((size_t)K * D * 2);
  float*  embn = (float*)alloc((size_t)K * 4);
  __bf16* xh  = (__bf16*)alloc((size_t)N * D * 2);
  __bf16* xl  = (__bf16*)alloc((size_t)N * D * 2);
  int*   idxi = (int*)alloc((size_t)N * 4);

  // Output layout: [quantized N*D][indices N][encodings N*K], all float.
  float* qout   = (float*)d_out;
  float* idxout = qout + (size_t)N * D;
  float* enc    = idxout + N;

  // 1) bf16 hi/lo splits
  {
    int n = N * H;
    split_bf16_kernel<<<(n + 255) / 256, 256, 0, stream>>>(features, fh, fl, n);
    n = D * H;
    split_bf16_kernel<<<(n + 255) / 256, 256, 0, stream>>>(Wp, wph, wpl, n);
    n = K * D;
    split_bf16_kernel<<<(n + 255) / 256, 256, 0, stream>>>(emb, eh, el, n);
  }

  // 2) codebook row norms
  rownorm_kernel<<<K, D, 0, stream>>>(emb, embn, D);

  // 3) fused projection + LayerNorm
  proj_ln_kernel<<<N / 16, 512, 0, stream>>>(fh, fl, wph, wpl, bp, gamma, beta,
                                             xh, xl, H, D);

  // 4) distance + argmin + gather
  dist_argmin_kernel<<<N / 32, 512, 0, stream>>>(xh, xl, eh, el, embn, emb,
                                                 qout, idxout, idxi, D, K);

  // 5) one-hot encodings
  {
    int kshift = 0;
    while ((1 << kshift) < K) kshift++;
    long long total4 = (long long)N * K / 4;
    long long grid = (total4 + 255) / 256;
    encodings_kernel<<<(unsigned)grid, 256, 0, stream>>>(idxi, enc, kshift, total4);
  }
}
